// GATv2Layer_18528488914947
// MI455X (gfx1250) — matrix operand order repaired and checked by reading, not compiled, not run on hardware
//
#include <hip/hip_runtime.h>
#include <hip/hip_bf16.h>

#define N_NODES 50000
#define N_EDGES 400000
#define IN_DIM 256
#define OUT_DIM 64
#define HEADS 2
#define ATTN_HIDDEN 64
#define ALPHA 0.2f
#define TEMP 0.55f

typedef __attribute__((ext_vector_type(16))) __bf16 v16bf;
typedef __attribute__((ext_vector_type(8)))  float  v8f;

union Frag {
    unsigned u[8];
    uint4    q[2];
    v16bf    v;
};

// Cheap f32x2 -> packed bf16x2: round-half-up + byte-perm (3 VALU ops total)
static __device__ __forceinline__ unsigned pack_bf16(float a, float b) {
    unsigned ua = __float_as_uint(a) + 0x8000u;
    unsigned ub = __float_as_uint(b) + 0x8000u;
    // result byte0,1 = ua bytes 2,3 ; byte2,3 = ub bytes 2,3
    return __builtin_amdgcn_perm(ub, ua, 0x07060302u);
}
static __device__ __forceinline__ unsigned short bf16_1(float a) {
    return (unsigned short)((__float_as_uint(a) + 0x8000u) >> 16);
}

// Monotone float <-> uint encoding for atomicMax on floats
static __device__ __forceinline__ unsigned enc_f32(float f) {
    unsigned u = __float_as_uint(f);
    return (u & 0x80000000u) ? ~u : (u | 0x80000000u);
}
static __device__ __forceinline__ float dec_f32(unsigned u) {
    u = (u & 0x80000000u) ? (u & 0x7FFFFFFFu) : ~u;
    return __uint_as_float(u);
}

// ---------------------------------------------------------------------------
// Prepack W [256][128] (f32) into bf16 B-fragment order:
// wpack[((kk*8 + nt)*32 + lane)*8 + v]; col = nt*16+(lane&15),
// K0 = kk*32 + (lane>>4)*16 + 2v  (contiguous k order, same as the A fragments).
// ---------------------------------------------------------------------------
__global__ void k_pack_w(const float* __restrict__ W, unsigned* __restrict__ wpack) {
    int t = blockIdx.x * blockDim.x + threadIdx.x;
    if (t >= 8 * 8 * 32 * 8) return;
    int v    = t & 7;
    int lane = (t >> 3) & 31;
    int nt   = (t >> 8) & 7;
    int kk   = (t >> 11) & 7;
    int col  = nt * 16 + (lane & 15);
    int K0   = kk * 32 + ((lane >> 4) * 16) + 2 * v;
    wpack[t] = pack_bf16(W[(size_t)K0 * 128 + col], W[(size_t)(K0 + 1) * 128 + col]);
}

// Prepack W_attn [2][128][64]: wapack[(((head*4+kk)*4+nt)*32+lane)*8+v]
__global__ void k_pack_wa(const float* __restrict__ Wa, unsigned* __restrict__ wapack) {
    int t = blockIdx.x * blockDim.x + threadIdx.x;
    if (t >= 2 * 4 * 4 * 32 * 8) return;
    int v    = t & 7;
    int lane = (t >> 3) & 31;
    int nt   = (t >> 8) & 3;
    int kk   = (t >> 10) & 3;
    int head = (t >> 12) & 1;
    int col  = nt * 16 + (lane & 15);
    int K0   = kk * 32 + ((lane >> 4) * 16) + 2 * v;
    const float* base = Wa + (size_t)head * 128 * 64;
    wapack[t] = pack_bf16(base[(size_t)K0 * 64 + col], base[(size_t)(K0 + 1) * 64 + col]);
}

// Convert x [50000][256] f32 -> row-major bf16 (one-shot, memory bound)
__global__ void k_cvt_x(const float* __restrict__ x, unsigned* __restrict__ xb) {
    int t = blockIdx.x * blockDim.x + threadIdx.x;      // one thread per 4 floats
    if (t >= N_NODES * IN_DIM / 4) return;
    float4 f = reinterpret_cast<const float4*>(x)[t];
    uint2 o;
    o.x = pack_bf16(f.x, f.y);
    o.y = pack_bf16(f.z, f.w);
    reinterpret_cast<uint2*>(xb)[t] = o;
}

// ---------------------------------------------------------------------------
// Kernel 1: Wh[n][128] = x[n][256] @ W[256][128], bf16 WMMA, f32 accumulate.
// 128 threads = 4 waves; block handles 16 rows x 128 cols; wave -> 16x32 tile.
// All fragment loads are raw 16B loads (operands pre-converted to bf16).
// ---------------------------------------------------------------------------
__global__ void __launch_bounds__(128) k_node_proj(const unsigned short* __restrict__ xb,
                                                   const unsigned* __restrict__ wpack,
                                                   float* __restrict__ Wh,
                                                   unsigned short* __restrict__ Whb) {
    const int lane    = threadIdx.x & 31;
    const int wave    = threadIdx.x >> 5;
    const int rowBase = blockIdx.x * 16;            // 3125 blocks * 16 = 50000 exactly
    const int r       = lane & 15;
    const int kq      = (lane >> 4) * 16;           // K-half offset for A fragments
    const unsigned short* xrow = xb + (size_t)(rowBase + r) * IN_DIM;

    v8f acc0 = {}, acc1 = {};
    const int nt0 = wave * 2;

#pragma unroll
    for (int kk = 0; kk < 8; ++kk) {
        // A fragment (16-bit A 16x32): q[0] = K kq..kq+7, q[1] = K kq+8..kq+15
        Frag a;
        a.q[0] = *reinterpret_cast<const uint4*>(xrow + kk * 32 + kq);
        a.q[1] = *reinterpret_cast<const uint4*>(xrow + kk * 32 + kq + 8);

        Frag b0, b1;
        const uint4* q0 = reinterpret_cast<const uint4*>(wpack + ((size_t)(kk * 8 + nt0) * 32 + lane) * 8);
        b0.q[0] = q0[0]; b0.q[1] = q0[1];
        const uint4* q1 = reinterpret_cast<const uint4*>(wpack + ((size_t)(kk * 8 + nt0 + 1) * 32 + lane) * 8);
        b1.q[0] = q1[0]; b1.q[1] = q1[1];

        acc0 = __builtin_amdgcn_wmma_f32_16x16x32_bf16(false, a.v, false, b0.v, (short)0, acc0, false, false);
        acc1 = __builtin_amdgcn_wmma_f32_16x16x32_bf16(false, a.v, false, b1.v, (short)0, acc1, false, false);
    }

    // C/D layout: VGPR v, lane l -> M = v + 8*(l>>4), N = (l&15) + nt*16
    const int mbase = (lane >> 4) * 8;
    const int c0    = nt0 * 16 + (lane & 15);
#pragma unroll
    for (int v = 0; v < 8; ++v) {
        const size_t m = (size_t)(rowBase + mbase + v);
        Wh[m * 128 + c0]       = acc0[v];
        Wh[m * 128 + c0 + 16]  = acc1[v];
        Whb[m * 128 + c0]      = bf16_1(acc0[v]);
        Whb[m * 128 + c0 + 16] = bf16_1(acc1[v]);
    }
}

// ---------------------------------------------------------------------------
// Kernel 2: per-edge attention logits via WMMA (dominant 13.1 GFLOP stage).
// One wave per (16-edge tile, head). K=128 = 4 steps: kk<2 -> Whb[src] half,
// kk>=2 -> Whb[dst] half (concat never straddles a K-step). Pure 16B loads.
// ---------------------------------------------------------------------------
__global__ void __launch_bounds__(256) k_edge_attn(const unsigned short* __restrict__ Whb,
                                                   const unsigned* __restrict__ wapack,
                                                   const float* __restrict__ a_vec,
                                                   const int* __restrict__ src,
                                                   const int* __restrict__ dst,
                                                   float* __restrict__ logits) {
    const int lane  = threadIdx.x & 31;
    const int waveG = blockIdx.x * 8 + (threadIdx.x >> 5);   // 6250*8 = 50000 waves
    const int tile  = waveG >> 1;                            // 25000 tiles exactly
    const int head  = waveG & 1;
    if (tile >= N_EDGES / 16) return;

    const int r  = lane & 15;
    const int kq = (lane >> 4) * 16;
    const int e0 = tile * 16 + r;
    const int sn = src[e0];
    const int dn = dst[e0];

    v8f acc[4] = {v8f{}, v8f{}, v8f{}, v8f{}};

#pragma unroll
    for (int kk = 0; kk < 4; ++kk) {
        const int node = (kk < 2) ? sn : dn;
        const unsigned short* zrow = Whb + (size_t)node * 128 + head * 64 + (kk & 1) * 32;

        Frag a;
        a.q[0] = *reinterpret_cast<const uint4*>(zrow + kq);
        a.q[1] = *reinterpret_cast<const uint4*>(zrow + kq + 8);

#pragma unroll
        for (int nt = 0; nt < 4; ++nt) {
            Frag b;
            const uint4* q = reinterpret_cast<const uint4*>(
                wapack + ((size_t)(((head * 4 + kk) * 4 + nt) * 32 + lane)) * 8);
            b.q[0] = q[0]; b.q[1] = q[1];
            acc[nt] = __builtin_amdgcn_wmma_f32_16x16x32_bf16(false, a.v, false, b.v, (short)0, acc[nt], false, false);
        }
    }

    // leaky_relu + dot with a_vec (N = nt*16 + (lane&15)), then half-wave reduce
    float av[4];
#pragma unroll
    for (int nt = 0; nt < 4; ++nt) av[nt] = a_vec[head * ATTN_HIDDEN + nt * 16 + (lane & 15)];

    const int mhalf = (lane >> 4) * 8;
#pragma unroll
    for (int v = 0; v < 8; ++v) {
        float p = 0.f;
#pragma unroll
        for (int nt = 0; nt < 4; ++nt) {
            float u = acc[nt][v];
            u = (u > 0.f) ? u : ALPHA * u;
            p += u * av[nt];
        }
        p += __shfl_xor(p, 1, 32);
        p += __shfl_xor(p, 2, 32);
        p += __shfl_xor(p, 4, 32);
        p += __shfl_xor(p, 8, 32);
        if ((lane & 15) == 0) {
            logits[(size_t)(tile * 16 + mhalf + v) * HEADS + head] = p * (1.0f / TEMP);
        }
    }
}

// ---------------------------------------------------------------------------
// Softmax / scatter stage
// ---------------------------------------------------------------------------
__global__ void k_init(unsigned* __restrict__ seg_max, float* __restrict__ seg_sum,
                       float* __restrict__ out) {
    int t = blockIdx.x * blockDim.x + threadIdx.x;
    if (t < N_NODES * HEADS) {
        seg_max[t] = enc_f32(-__builtin_inff());
        seg_sum[t] = 0.f;
    }
    if (t < N_NODES * OUT_DIM) out[t] = 0.f;
}

__global__ void k_segmax(const float* __restrict__ logits, const int* __restrict__ dst,
                         unsigned* __restrict__ seg_max) {
    int t = blockIdx.x * blockDim.x + threadIdx.x;
    if (t >= N_EDGES * HEADS) return;
    int e = t >> 1, h = t & 1;
    atomicMax(&seg_max[dst[e] * HEADS + h], enc_f32(logits[t]));
}

__global__ void k_expsum(float* __restrict__ logits, const int* __restrict__ dst,
                         const unsigned* __restrict__ seg_max, float* __restrict__ seg_sum) {
    int t = blockIdx.x * blockDim.x + threadIdx.x;
    if (t >= N_EDGES * HEADS) return;
    int e = t >> 1, h = t & 1;
    float m  = dec_f32(seg_max[dst[e] * HEADS + h]);
    float ex = __expf(logits[t] - m);
    logits[t] = ex;
    atomicAdd(&seg_sum[dst[e] * HEADS + h], ex);
}

__global__ void k_scatter(const float* __restrict__ Wh, const float* __restrict__ ex,
                          const float* __restrict__ seg_sum,
                          const int* __restrict__ src, const int* __restrict__ dst,
                          float* __restrict__ out) {
    int t = blockIdx.x * blockDim.x + threadIdx.x;
    if (t >= N_EDGES * OUT_DIM) return;
    int e = t >> 6, f = t & 63;
    int d = dst[e], s = src[e];
    float a0 = ex[(size_t)e * 2 + 0] / (seg_sum[(size_t)d * 2 + 0] + 1e-9f);
    float a1 = ex[(size_t)e * 2 + 1] / (seg_sum[(size_t)d * 2 + 1] + 1e-9f);
    float msg = a0 * Wh[(size_t)s * 128 + f] + a1 * Wh[(size_t)s * 128 + 64 + f];
    atomicAdd(&out[(size_t)d * OUT_DIM + f], 0.5f * msg);   // mean over 2 heads
}

// ---------------------------------------------------------------------------
extern "C" void kernel_launch(void* const* d_in, const int* in_sizes, int n_in,
                              void* d_out, int out_size, void* d_ws, size_t ws_size,
                              hipStream_t stream) {
    const float* x      = (const float*)d_in[0];   // [50000,256]
    const float* W      = (const float*)d_in[1];   // [256,2,64]
    const float* Wa     = (const float*)d_in[2];   // [2,128,64]
    const float* a_vec  = (const float*)d_in[3];   // [2,64]
    const int*   src    = (const int*)d_in[4];     // [400000]
    const int*   dst    = (const int*)d_in[5];     // [400000]
    float*       out    = (float*)d_out;           // [50000,64]

    char* ws = (char*)d_ws;
    size_t o = 0;
    float*          Wh      = (float*)(ws + o);          o += (size_t)N_NODES * 128 * 4;   // 25.6 MB
    unsigned short* Whb     = (unsigned short*)(ws + o); o += (size_t)N_NODES * 128 * 2;   // 12.8 MB
    unsigned short* xb      = (unsigned short*)(ws + o); o += (size_t)N_NODES * IN_DIM * 2;// 25.6 MB
    unsigned*       wpack   = (unsigned*)(ws + o);       o += 65536;
    unsigned*       wapack  = (unsigned*)(ws + o);       o += 32768;
    float*          logits  = (float*)(ws + o);          o += (size_t)N_EDGES * HEADS * 4; // 3.2 MB
    unsigned*       seg_max = (unsigned*)(ws + o);       o += (size_t)N_NODES * HEADS * 4;
    float*          seg_sum = (float*)(ws + o);          o += (size_t)N_NODES * HEADS * 4;

    k_pack_w  <<<(16384 + 255) / 256, 256, 0, stream>>>(W, wpack);
    k_pack_wa <<<(8192  + 255) / 256, 256, 0, stream>>>(Wa, wapack);
    k_cvt_x   <<<(N_NODES * IN_DIM / 4 + 255) / 256, 256, 0, stream>>>(x, (unsigned*)xb);
    k_init    <<<(N_NODES * OUT_DIM + 255) / 256, 256, 0, stream>>>(seg_max, seg_sum, out);

    k_node_proj<<<N_NODES / 16, 128, 0, stream>>>(xb, wpack, Wh, Whb);                 // 3125 blocks
    k_edge_attn<<<(N_EDGES / 16) * HEADS / 8, 256, 0, stream>>>(Whb, wapack, a_vec,    // 6250 blocks
                                                                src, dst, logits);

    k_segmax <<<(N_EDGES * HEADS + 255) / 256, 256, 0, stream>>>(logits, dst, seg_max);
    k_expsum <<<(N_EDGES * HEADS + 255) / 256, 256, 0, stream>>>(logits, dst, seg_max, seg_sum);
    k_scatter<<<(N_EDGES * OUT_DIM + 255) / 256, 256, 0, stream>>>(Wh, logits, seg_sum,
                                                                   src, dst, out);
}
